// SeqAttnContext_35493609734382
// MI455X (gfx1250) — matrix operand order repaired and checked by reading, not compiled, not run on hardware
//
#include <hip/hip_runtime.h>
#include <hip/hip_bf16.h>

// ---------------------------------------------------------------------------
// SeqAttnContext on MI455X (gfx1250), bf16 WMMA + fused flash-attention.
// B=64, LX=LY=1024, D=300 (padded to DP=320 = 10 K-tiles of 32).
// ---------------------------------------------------------------------------

#define B_   64
#define LX_  1024
#define LY_  1024
#define D_   300
#define DP   320
#define KT   (DP / 32)   // 10 K-tiles for the bf16 WMMA (K=32)
#define NT_P (DP / 16)   // 20 N-tiles for projection output
#define NT_C 19          // context N-tiles: 19*16 = 304 >= 300

typedef __bf16 bf16;
typedef __attribute__((ext_vector_type(16))) __bf16 v16bf;
typedef __attribute__((ext_vector_type(8)))  __bf16 v8bf;
typedef __attribute__((ext_vector_type(8)))  float  v8f;
typedef __attribute__((ext_vector_type(4)))  float  v4f;

union frag16 { v16bf v; v8bf h[2]; };

__device__ __forceinline__ bf16 f2bf(float f) {
  unsigned u = __builtin_bit_cast(unsigned, f);
  unsigned r = (u + 0x7fffu + ((u >> 16) & 1u)) >> 16;
  return __builtin_bit_cast(bf16, (unsigned short)r);
}

// XOR-shuffle within the 16-lane half (matches C/D frag: rows in VGPR idx,
// cols in lanes; masks 1,2,4,8 never cross the half boundary).
#define SWZF(v, m) __builtin_bit_cast(float,                                   \
    __builtin_amdgcn_ds_swizzle(__builtin_bit_cast(int, (v)),                  \
                                (((m) << 10) | 0x1f)))

__device__ __forceinline__ float halfmax(float v) {
  v = fmaxf(v, SWZF(v, 1)); v = fmaxf(v, SWZF(v, 2));
  v = fmaxf(v, SWZF(v, 4)); v = fmaxf(v, SWZF(v, 8));
  return v;
}
__device__ __forceinline__ float halfsum(float v) {
  v += SWZF(v, 1); v += SWZF(v, 2); v += SWZF(v, 4); v += SWZF(v, 8);
  return v;
}

// 16-bit A-matrix frag (16x32): lane L holds row L&15; half h holds K chunks
// [h*8, h*8+8) and [16+h*8, 16+h*8+8).
__device__ __forceinline__ v16bf load_a_frag(const bf16* rowbase, int stride,
                                             int l15, int half, int kbase) {
  frag16 u;
  const bf16* p = rowbase + (size_t)l15 * stride + kbase + half * 8;
  u.h[0] = *(const v8bf*)(p);
  u.h[1] = *(const v8bf*)(p + 16);
  return u.v;
}

// 16-bit B-matrix frag (32x16): lane L holds column L&15; half h holds
// K chunks [h*8, h*8+8) and [16+h*8, 16+h*8+8) from colptr.
__device__ __forceinline__ v16bf load_b_frag(const bf16* colptr, int half) {
  frag16 u;
  const bf16* p = colptr + half * 8;
  u.h[0] = *(const v8bf*)(p);
  u.h[1] = *(const v8bf*)(p + 16);
  return u.v;
}

// ---------------------------------------------------------------------------
// Kernel 1: W (300x300 f32, row-major [e][d]) -> Wb (320x320 bf16, zero-pad)
//           b (300 f32) -> biasp (320 f32, zero-pad)
// Wb[e][d] is directly the B-matrix source for proj GEMM (lane = e, k = d).
// ---------------------------------------------------------------------------
__global__ __launch_bounds__(256) void prep_w_kernel(
    const float* __restrict__ W, const float* __restrict__ bvec,
    bf16* __restrict__ Wb, float* __restrict__ biasp) {
  int i = blockIdx.x * 256 + threadIdx.x;
  if (i < DP * DP) {
    int e = i / DP, d = i - e * DP;
    float v = (e < D_ && d < D_) ? W[e * D_ + d] : 0.0f;
    Wb[i] = f2bf(v);
  }
  if (i < DP) biasp[i] = (i < D_) ? bvec[i] : 0.0f;
}

// ---------------------------------------------------------------------------
// Kernel 2: tiled transpose y (B,LY,300 f32) -> yT (B,DP,LY bf16), zero-pad d.
// yT[b][d][t] is the B-matrix source for the context GEMM (lane = d, k = t).
// ---------------------------------------------------------------------------
__global__ __launch_bounds__(256) void prep_yT_kernel(
    const float* __restrict__ y, bf16* __restrict__ yT) {
  __shared__ bf16 tile[32][33];
  int blk = blockIdx.x;
  int dt = blk % (DP / 32);          // 10 d-tiles
  int tt = (blk / (DP / 32)) % (LY_ / 32);
  int b  = blk / ((DP / 32) * (LY_ / 32));
  int lx = threadIdx.x & 31;
  int ly = threadIdx.x >> 5;         // 8 rows per pass
#pragma unroll
  for (int i = 0; i < 32; i += 8) {
    int t = tt * 32 + ly + i;
    int d = dt * 32 + lx;
    float v = (d < D_) ? y[((size_t)b * LY_ + t) * D_ + d] : 0.0f;
    tile[ly + i][lx] = f2bf(v);
  }
  __syncthreads();
#pragma unroll
  for (int i = 0; i < 32; i += 8) {
    int d = dt * 32 + ly + i;
    int t = tt * 32 + lx;
    yT[((size_t)b * DP + d) * LY_ + t] = tile[lx][ly + i];
  }
}

// ---------------------------------------------------------------------------
// Kernel 3: shared projection  proj = relu(in @ W^T + b), bf16 out, DP-padded.
// Rows [0,65536) come from x -> x_proj ; rows [65536,131072) from y -> y_proj.
// 4 waves/block, 16 rows/wave, N=320 (20 tiles), K=320 (10 WMMA steps).
// ---------------------------------------------------------------------------
__global__ __launch_bounds__(128) void proj_kernel(
    const float* __restrict__ x, const float* __restrict__ y,
    const bf16* __restrict__ Wb, const float* __restrict__ biasp,
    bf16* __restrict__ xproj, bf16* __restrict__ yproj) {
  __shared__ bf16 abuf[4][16 * DP];        // 40 KB: per-wave A staging
  const int wave = threadIdx.x >> 5;
  const int lane = threadIdx.x & 31;
  const int half = lane >> 4;
  const int l15  = lane & 15;

  size_t gr0 = (size_t)blockIdx.x * 64 + (size_t)wave * 16;  // wave's 1st row
  const bool isx = gr0 < (size_t)B_ * LX_;
  const size_t ridx = isx ? gr0 : gr0 - (size_t)B_ * LX_;
  const float* src = (isx ? x : y) + ridx * D_;
  bf16* dst = (isx ? xproj : yproj) + ridx * DP;

  // stage 16 input rows as bf16 into LDS, vectorized:
  // 300 f32 = 75 aligned float4 chunks (row base 1200 B, 16B aligned).
  bf16* ab = abuf[wave];
  for (int row = 0; row < 16; ++row) {
    const float* srow = src + (size_t)row * D_;
    bf16* arow = ab + row * DP;
#pragma unroll
    for (int q = lane; q < 75; q += 32) {
      v4f f = *(const v4f*)(srow + q * 4);
      union { bf16 b[4]; unsigned long long u; } pk;
      pk.b[0] = f2bf(f[0]); pk.b[1] = f2bf(f[1]);
      pk.b[2] = f2bf(f[2]); pk.b[3] = f2bf(f[3]);
      *(unsigned long long*)(arow + q * 4) = pk.u;   // ds_store_b64
    }
    if (lane < 10) *(int*)(arow + D_ + lane * 2) = 0;  // zero-pad 300..319
  }
  asm volatile("s_wait_dscnt 0" ::: "memory");

  // resident A-fragments (10 x 8 VGPRs)
  v16bf a[KT];
#pragma unroll
  for (int kt = 0; kt < KT; ++kt)
    a[kt] = load_a_frag(ab, DP, l15, half, kt * 32);

  for (int nt = 0; nt < NT_P; ++nt) {
    float bb = biasp[nt * 16 + l15];
    v8f c;
#pragma unroll
    for (int i = 0; i < 8; ++i) c[i] = bb;   // bias broadcast into C
#pragma unroll
    for (int kt = 0; kt < KT; ++kt) {
      v16bf bfrag = load_b_frag(Wb + (size_t)(nt * 16 + l15) * DP + kt * 32, half);
      c = __builtin_amdgcn_wmma_f32_16x16x32_bf16(false, a[kt], false, bfrag,
                                                  (short)0, c, false, false);
    }
#pragma unroll
    for (int v = 0; v < 8; ++v) {            // fused ReLU + bf16 store
      float val = fmaxf(c[v], 0.0f);
      dst[(size_t)(v + 8 * half) * DP + nt * 16 + l15] = f2bf(val);
    }
  }
}

// ---------------------------------------------------------------------------
// Kernel 4: fused flash attention.
// Block = 4 waves x 16 x-rows = 64 rows; grid = B * LX/64 = 1024 blocks.
// Per 32-column y-step: 20 score WMMAs (K=320, two independent accumulation
// chains interleaved), mask->-inf, online softmax (ds_swizzle reductions),
// P transposed via 1KB/wave LDS, 19 context WMMAs against yT (K=32).
// ---------------------------------------------------------------------------
__global__ __launch_bounds__(128) void attn_kernel(
    const bf16* __restrict__ xproj, const bf16* __restrict__ yproj,
    const bf16* __restrict__ yT, const unsigned char* __restrict__ ymask,
    float* __restrict__ out) {
  __shared__ bf16 pbuf[4][16 * 32];        // per-wave P staging (4 KB)
  const int wave = threadIdx.x >> 5;
  const int lane = threadIdx.x & 31;
  const int half = lane >> 4;
  const int l15  = lane & 15;

  const int b      = blockIdx.x >> 4;      // 16 row-blocks per batch
  const int rowblk = blockIdx.x & 15;
  const int r0     = rowblk * 64 + wave * 16;

  const bf16* xp  = xproj + ((size_t)b * LX_ + r0) * DP;
  const bf16* ypb = yproj + (size_t)b * LY_ * DP;
  const bf16* yTb = yT    + (size_t)b * DP * LY_;
  const unsigned char* mk = ymask + (size_t)b * LY_;

  // resident x_proj A-fragments for this wave's 16 rows
  v16bf xa[KT];
#pragma unroll
  for (int kt = 0; kt < KT; ++kt)
    xa[kt] = load_a_frag(xp, DP, l15, half, kt * 32);

  const float NEGINF = -__builtin_inff();
  float m[8], l[8];
  v8f acc[NT_C];
#pragma unroll
  for (int v = 0; v < 8; ++v) { m[v] = NEGINF; l[v] = 0.0f; }
#pragma unroll
  for (int nt = 0; nt < NT_C; ++nt)
#pragma unroll
    for (int v = 0; v < 8; ++v) acc[nt][v] = 0.0f;

  bf16* pb = pbuf[wave];

  for (int t0 = 0; t0 < LY_; t0 += 32) {
    // prefetch next y_proj tile: one cacheline per lane covers 32 rows
    if (t0 + 32 < LY_)
      __builtin_prefetch(ypb + (size_t)(t0 + 32 + lane) * DP, 0, 0);

    // ---- scores: two 16-col subtiles, K=320, interleaved chains ----
    v8f S0, S1;
#pragma unroll
    for (int i = 0; i < 8; ++i) { S0[i] = 0.0f; S1[i] = 0.0f; }
    const int tA = t0 + l15;             // this lane's column, subtile 0
    const int tB = t0 + 16 + l15;        // subtile 1
    const bf16* pA = ypb + (size_t)tA * DP;
    const bf16* pB = ypb + (size_t)tB * DP;
#pragma unroll
    for (int kt = 0; kt < KT; ++kt) {
      v16bf bA = load_b_frag(pA + kt * 32, half);
      v16bf bB = load_b_frag(pB + kt * 32, half);
      S0 = __builtin_amdgcn_wmma_f32_16x16x32_bf16(false, xa[kt], false, bA,
                                                   (short)0, S0, false, false);
      S1 = __builtin_amdgcn_wmma_f32_16x16x32_bf16(false, xa[kt], false, bB,
                                                   (short)0, S1, false, false);
    }
    if (mk[tA]) {                         // padded column -> -inf
#pragma unroll
      for (int v = 0; v < 8; ++v) S0[v] = NEGINF;
    }
    if (mk[tB]) {
#pragma unroll
      for (int v = 0; v < 8; ++v) S1[v] = NEGINF;
    }

    // ---- online softmax update ----
    float corr[8], rs[8];
#pragma unroll
    for (int v = 0; v < 8; ++v) {
      float tm = halfmax(fmaxf(S0[v], S1[v]));
      float nm = fmaxf(m[v], tm);
      corr[v] = __expf(m[v] - nm);
      m[v] = nm;
      S0[v] = __expf(S0[v] - nm);
      S1[v] = __expf(S1[v] - nm);
      rs[v] = S0[v] + S1[v];
    }
#pragma unroll
    for (int v = 0; v < 8; ++v) l[v] = l[v] * corr[v] + halfsum(rs[v]);
#pragma unroll
    for (int nt = 0; nt < NT_C; ++nt)
#pragma unroll
      for (int v = 0; v < 8; ++v) acc[nt][v] *= corr[v];

    // ---- transpose P (C/D layout -> A layout) through LDS ----
#pragma unroll
    for (int v = 0; v < 8; ++v) {
      pb[(v + 8 * half) * 32 + l15]      = f2bf(S0[v]);
      pb[(v + 8 * half) * 32 + 16 + l15] = f2bf(S1[v]);
    }
    asm volatile("s_wait_dscnt 0" ::: "memory");
    v16bf pfrag = load_a_frag(pb, 32, l15, half, 0);
    asm volatile("" ::: "memory");

    // ---- context accumulation: acc += P (16x32) @ y (32 x 304) ----
#pragma unroll
    for (int nt = 0; nt < NT_C; ++nt) {
      v16bf bfrag = load_b_frag(yTb + (size_t)(nt * 16 + l15) * LY_ + t0, half);
      acc[nt] = __builtin_amdgcn_wmma_f32_16x16x32_bf16(false, pfrag, false, bfrag,
                                                        (short)0, acc[nt], false, false);
    }
  }

  // ---- epilogue: divide by softmax denominator, store f32 ----
  float inv[8];
#pragma unroll
  for (int v = 0; v < 8; ++v) inv[v] = 1.0f / l[v];
  float* ob = out + ((size_t)b * LX_ + r0) * D_;
#pragma unroll
  for (int nt = 0; nt < NT_C; ++nt) {
    int d = nt * 16 + l15;
    if (d < D_) {
#pragma unroll
      for (int v = 0; v < 8; ++v)
        ob[(size_t)(v + 8 * half) * D_ + d] = acc[nt][v] * inv[v];
    }
  }
}

// ---------------------------------------------------------------------------
// Launch. Inputs: x, y, y_mask(bool), W, b. Output: (B,LX,300) f32.
// Workspace: Wb 200KB + biasp 1.25KB + x_proj/y_proj/yT (3 x 40MB) ~ 126 MB.
// ---------------------------------------------------------------------------
extern "C" void kernel_launch(void* const* d_in, const int* in_sizes, int n_in,
                              void* d_out, int out_size, void* d_ws, size_t ws_size,
                              hipStream_t stream) {
  const float* x  = (const float*)d_in[0];
  const float* y  = (const float*)d_in[1];
  const unsigned char* ym = (const unsigned char*)d_in[2];
  const float* W  = (const float*)d_in[3];
  const float* bv = (const float*)d_in[4];
  float* out = (float*)d_out;

  char* ws = (char*)d_ws;
  size_t off = 0;
  bf16* Wb = (bf16*)(ws + off);      off += (size_t)DP * DP * sizeof(bf16);
  float* biasp = (float*)(ws + off); off += (size_t)DP * sizeof(float);
  bf16* xproj = (bf16*)(ws + off);   off += (size_t)B_ * LX_ * DP * sizeof(bf16);
  bf16* yproj = (bf16*)(ws + off);   off += (size_t)B_ * LY_ * DP * sizeof(bf16);
  bf16* yT = (bf16*)(ws + off);      off += (size_t)B_ * DP * LY_ * sizeof(bf16);
  (void)ws_size; (void)in_sizes; (void)n_in; (void)out_size;

  // 1) pack W -> bf16 (padded) + padded bias
  prep_w_kernel<<<(DP * DP + 255) / 256, 256, 0, stream>>>(W, bv, Wb, biasp);

  // 2) transpose y -> bf16 yT[b][d][t]
  prep_yT_kernel<<<B_ * (DP / 32) * (LY_ / 32), 256, 0, stream>>>(y, yT);

  // 3) shared projection (x and y rows in one grid): 131072 rows / 64 per block
  proj_kernel<<<(B_ * (LX_ + LY_)) / 64, 128, 0, stream>>>(x, y, Wb, biasp,
                                                           xproj, yproj);

  // 4) fused masked-softmax attention
  attn_kernel<<<B_ * (LX_ / 64), 128, 0, stream>>>(xproj, yproj, yT, ym, out);
}
